// FKAN_91044716741314
// MI455X (gfx1250) — compile-verified, matrix operand order checked by reading, not run on hardware
//
#include <hip/hip_runtime.h>
#include <hip/hip_fp16.h>

// ---------------- problem constants ----------------
#define NB    32768          // batch
#define LDIM  512            // latent width
#define KH    8              // harmonics
#define FPI   16             // features per input (8 sin + 8 cos)
#define KDIM  (LDIM * FPI)   // 8192 GEMM reduction dim

// ---------------- GEMM tiling ----------------
#define BM   128
#define BN   128
#define BK   64              // 4 inputs worth of features per K-step
#define LDT  72              // padded LDS row stride (halves): 64 + 8
#define NKT  (KDIM / BK)     // 128 K-iterations
#define BUFH ((BM + BN) * LDT)          // halves per double-buffer slot
#define SMEM_BYTES (2 * BUFH * 2)       // 73,728 B dynamic LDS

typedef __attribute__((ext_vector_type(16))) _Float16 v16h;
typedef __attribute__((ext_vector_type(8)))  float    v8f;

union V16U { uint4 u[2]; v16h v; };

// sin(k*x), cos(k*x) for k=0..7 via one sincos + angle-addition recurrence.
__device__ __forceinline__ void feat_recur(float x, float* s, float* c) {
    float s1, c1;
    __sincosf(x, &s1, &c1);
    s[0] = 0.0f; c[0] = 1.0f;
    s[1] = s1;   c[1] = c1;
#pragma unroll
    for (int k = 2; k < KH; ++k) {
        s[k] = s[k - 1] * c1 + c[k - 1] * s1;
        c[k] = c[k - 1] * c1 - s[k - 1] * s1;
    }
}

// Write the 16 f16 features of one scalar activation to LDS (32 contiguous bytes).
__device__ __forceinline__ void write_features(float x, __half2* dst) {
    float s[KH], c[KH];
    feat_recur(x, s, c);
#pragma unroll
    for (int j = 0; j < 4; ++j) dst[j]     = __floats2half2_rn(s[2 * j], s[2 * j + 1]);
#pragma unroll
    for (int j = 0; j < 4; ++j) dst[4 + j] = __floats2half2_rn(c[2 * j], c[2 * j + 1]);
}

// Fill one double-buffer slot: A tile = on-the-fly sincos features (VALU + ds),
// B tile = packed weights streamed with async global->LDS (ASYNCcnt path).
__device__ __forceinline__ void fill_tiles(const float* __restrict__ hin,
                                           const __half* __restrict__ Wp,
                                           __half* At, __half* Bt,
                                           int bm0, int bn0, int k0, int tid) {
    const int i0 = k0 / FPI;
#pragma unroll
    for (int r = 0; r < 2; ++r) {
        int q   = tid + r * 256;       // 0..511 : 128 rows x 4 inputs
        int row = q >> 2;
        int il  = q & 3;
        float xv = hin[(size_t)(bm0 + row) * LDIM + (i0 + il)];
        write_features(xv, (__half2*)(At + row * LDT + il * FPI));
    }
    // 64 B per thread, LDS offset = low 32 bits of the generic shared pointer.
    int n  = tid >> 1;
    int hs = tid & 1;
    unsigned int lds = (unsigned int)(uintptr_t)(Bt + n * LDT + hs * 32);
    unsigned long long ga =
        (unsigned long long)(uintptr_t)(Wp + (size_t)(bn0 + n) * KDIM + k0 + hs * 32);
    asm volatile(
        "global_load_async_to_lds_b128 %0, %1, off\n\t"
        "global_load_async_to_lds_b128 %0, %1, off offset:16\n\t"
        "global_load_async_to_lds_b128 %0, %1, off offset:32\n\t"
        "global_load_async_to_lds_b128 %0, %1, off offset:48"
        :: "v"(lds), "v"(ga) : "memory");
}

// ---------------------------------------------------------------------------
// Repack A,B (f32, [out, LDIM, KH]) into Wp (f16, [out, KDIM]) with
// Wp[o][i*16 + k] = A[o,i,k], Wp[o][i*16 + 8 + k] = B[o,i,k].
// ---------------------------------------------------------------------------
__global__ void pack_w(const float* __restrict__ A, const float* __restrict__ B,
                       __half* __restrict__ Wp, int out) {
    int idx = blockIdx.x * blockDim.x + threadIdx.x;
    if (idx >= out * LDIM) return;
    int o = idx >> 9, i = idx & (LDIM - 1);
    const float* a = A + (size_t)idx * KH;
    const float* b = B + (size_t)idx * KH;
    __half* w = Wp + (size_t)o * KDIM + i * FPI;
#pragma unroll
    for (int k = 0; k < KH; ++k) {
        w[k]      = __float2half_rn(a[k]);
        w[KH + k] = __float2half_rn(b[k]);
    }
}

// ---------------------------------------------------------------------------
// Layer 0: in=1.  h0[b,o] = sum_k A0[o,0,k] sin(k x_b) + B0[o,0,k] cos(k x_b)
// ---------------------------------------------------------------------------
__global__ __launch_bounds__(256) void fkan_layer0(const float* __restrict__ x,
                                                   const float* __restrict__ A0,
                                                   const float* __restrict__ B0,
                                                   float* __restrict__ h0) {
    int idx = blockIdx.x * 256 + threadIdx.x;     // NB * 32 threads total
    int b = idx >> 5;
    int o0 = (idx & 31) << 4;
    float s[KH], c[KH];
    feat_recur(x[b], s, c);
#pragma unroll
    for (int oo = 0; oo < 16; ++oo) {
        int o = o0 + oo;
        const float* a = A0 + o * KH;
        const float* w = B0 + o * KH;
        float acc = 0.0f;
#pragma unroll
        for (int k = 0; k < KH; ++k)
            acc = fmaf(a[k], s[k], fmaf(w[k], c[k], acc));
        h0[(size_t)b * LDIM + o] = acc;
    }
}

// ---------------------------------------------------------------------------
// Middle layers: fused featurize + WMMA GEMM, double-buffered LDS,
// one barrier per K-iteration, async weight staging.
//   hout[NB, LDIM] = Feat(hin)[NB, KDIM] @ Wp^T[KDIM, LDIM]
// 256 threads = 8 waves; wave owns a 64x32 sub-tile (4x2 v8f accumulators).
// ---------------------------------------------------------------------------
__global__ __launch_bounds__(256) void fkan_gemm(const float* __restrict__ hin,
                                                 const __half* __restrict__ Wp,
                                                 float* __restrict__ hout) {
    extern __shared__ __half smem[];   // [2][ At(128*72) | Bt(128*72) ]

    const int tid  = threadIdx.x;
    const int bm0  = blockIdx.x * BM;
    const int bn0  = blockIdx.y * BN;
    const int wid  = tid >> 5;
    const int lane = tid & 31;
    const int l15  = lane & 15;
    const int lhi  = lane >> 4;          // 0 | 1
    const int koff = lhi << 3;           // hi-half lanes take K+8 (16-bit A/B layout)
    const int mw   = (wid >> 2) * 64;    // wave M offset within tile
    const int nw   = (wid & 3) * 32;     // wave N offset within tile

    v8f acc[4][2];
#pragma unroll
    for (int mt = 0; mt < 4; ++mt)
#pragma unroll
        for (int nt = 0; nt < 2; ++nt)
#pragma unroll
            for (int r = 0; r < 8; ++r) acc[mt][nt][r] = 0.0f;

    // prologue: fill buffer 0
    fill_tiles(hin, Wp, smem, smem + BM * LDT, bm0, bn0, 0, tid);

    for (int kt = 0; kt < NKT; ++kt) {
        __half* At = smem + (kt & 1) * BUFH;
        __half* Bt = At + BM * LDT;

        // publish current buffer: async B landed + all waves' fills done
        asm volatile("s_wait_asynccnt 0x0" ::: "memory");
        __syncthreads();

        // prefetch next buffer while computing this one
        if (kt + 1 < NKT) {
            __half* Atn = smem + ((kt + 1) & 1) * BUFH;
            fill_tiles(hin, Wp, Atn, Atn + BM * LDT, bm0, bn0, (kt + 1) * BK, tid);
        }

        // ---- two 16x16x32 K-steps on the current buffer ----
#pragma unroll
        for (int kk = 0; kk < BK; kk += 32) {
            v16h a[4], b[2];
#pragma unroll
            for (int mt = 0; mt < 4; ++mt) {
                const __half* p = &At[(mw + mt * 16 + l15) * LDT + kk + koff];
                V16U t;
                t.u[0] = *(const uint4*)p;          // halves 0..7  -> K..K+7
                t.u[1] = *(const uint4*)(p + 16);   // halves 8..15 -> K+16..K+23
                a[mt] = t.v;
            }
#pragma unroll
            for (int nt = 0; nt < 2; ++nt) {
                const __half* p = &Bt[(nw + nt * 16 + l15) * LDT + kk + koff];
                V16U t;
                t.u[0] = *(const uint4*)p;
                t.u[1] = *(const uint4*)(p + 16);
                b[nt] = t.v;
            }
#pragma unroll
            for (int mt = 0; mt < 4; ++mt)
#pragma unroll
                for (int nt = 0; nt < 2; ++nt)
                    acc[mt][nt] = __builtin_amdgcn_wmma_f32_16x16x32_f16(
                        false, a[mt], false, b[nt], (short)0, acc[mt][nt],
                        false, false);
        }
    }

    // ---- epilogue: 32-bit C/D layout -> f32 activations for next layer ----
#pragma unroll
    for (int mt = 0; mt < 4; ++mt)
#pragma unroll
        for (int nt = 0; nt < 2; ++nt) {
            int col  = bn0 + nw + nt * 16 + l15;
            int row0 = bm0 + mw + mt * 16 + lhi * 8;
#pragma unroll
            for (int r = 0; r < 8; ++r)
                hout[(size_t)(row0 + r) * LDIM + col] = acc[mt][nt][r];
        }
}

// ---------------------------------------------------------------------------
// Layer 4: out=1.  One wave32 per batch row; shuffle reduction.
// ---------------------------------------------------------------------------
__global__ __launch_bounds__(256) void fkan_layer4(const float* __restrict__ hin,
                                                   const __half* __restrict__ Wp4,
                                                   float* __restrict__ y) {
    int wid  = threadIdx.x >> 5;
    int lane = threadIdx.x & 31;
    int b    = blockIdx.x * 8 + wid;
    float acc = 0.0f;
    for (int i = lane; i < LDIM; i += 32) {
        float s[KH], c[KH];
        feat_recur(hin[(size_t)b * LDIM + i], s, c);
        const __half* w = Wp4 + i * FPI;
#pragma unroll
        for (int k = 0; k < KH; ++k)
            acc = fmaf(s[k], __half2float(w[k]),
                  fmaf(c[k], __half2float(w[KH + k]), acc));
    }
#pragma unroll
    for (int off = 16; off > 0; off >>= 1)
        acc += __shfl_xor(acc, off, 32);
    if (lane == 0) y[b] = acc;
}

// ---------------------------------------------------------------------------
extern "C" void kernel_launch(void* const* d_in, const int* in_sizes, int n_in,
                              void* d_out, int out_size, void* d_ws, size_t ws_size,
                              hipStream_t stream) {
    const float* x  = (const float*)d_in[0];
    const float* A0 = (const float*)d_in[1];
    const float* B0 = (const float*)d_in[2];
    const float* A1 = (const float*)d_in[3];
    const float* B1 = (const float*)d_in[4];
    const float* A2 = (const float*)d_in[5];
    const float* B2 = (const float*)d_in[6];
    const float* A3 = (const float*)d_in[7];
    const float* B3 = (const float*)d_in[8];
    const float* A4 = (const float*)d_in[9];
    const float* B4 = (const float*)d_in[10];

    char* ws = (char*)d_ws;
    float*  hA = (float*)ws;                                   // 64 MB
    float*  hB = (float*)(ws + (size_t)NB * LDIM * 4);         // 64 MB
    __half* W1 = (__half*)(ws + (size_t)NB * LDIM * 8);        // 8 MB each
    __half* W2 = W1 + (size_t)LDIM * KDIM;
    __half* W3 = W2 + (size_t)LDIM * KDIM;
    __half* W4 = W3 + (size_t)LDIM * KDIM;                     // 16 KB

    // weight repack (f32 [o,i,k] -> f16 [o, i*16 + {k, 8+k}])
    pack_w<<<(LDIM * LDIM + 255) / 256, 256, 0, stream>>>(A1, B1, W1, LDIM);
    pack_w<<<(LDIM * LDIM + 255) / 256, 256, 0, stream>>>(A2, B2, W2, LDIM);
    pack_w<<<(LDIM * LDIM + 255) / 256, 256, 0, stream>>>(A3, B3, W3, LDIM);
    pack_w<<<(1 * LDIM + 255) / 256,    256, 0, stream>>>(A4, B4, W4, 1);

    // layer 0 (in = 1)
    fkan_layer0<<<(NB * 32) / 256, 256, 0, stream>>>(x, A0, B0, hA);

    // layers 1..3: fused featurize + WMMA GEMM, ping-pong activations
    dim3 g(NB / BM, LDIM / BN, 1);
    fkan_gemm<<<g, 256, SMEM_BYTES, stream>>>(hA, W1, hB);
    fkan_gemm<<<g, 256, SMEM_BYTES, stream>>>(hB, W2, hA);
    fkan_gemm<<<g, 256, SMEM_BYTES, stream>>>(hA, W3, hB);

    // layer 4 (out = 1)
    fkan_layer4<<<NB / 8, 256, 0, stream>>>(hB, W4, (float*)d_out);
}
